// SocialLSTMClassifier_62878321214260
// MI455X (gfx1250) — matrix operand order repaired and checked by reading, not compiled, not run on hardware
//
#include <hip/hip_runtime.h>
#include <hip/hip_bf16.h>

typedef __attribute__((ext_vector_type(16))) _Float16 v16h;
typedef __attribute__((ext_vector_type(8)))  _Float16 v8h;
typedef __attribute__((ext_vector_type(8)))  float    v8f;

namespace {
constexpr int kB  = 4096;   // batch
constexpr int kT  = 15;     // time steps
constexpr int kN  = 32;     // neighbors
constexpr int kF  = 16;     // input features
constexpr int kH  = 64;     // hidden
constexpr int kG  = 4 * kH; // 256 LSTM gate width
constexpr int kMB = 16;     // batches per block in LSTM kernel

// dynamic-LDS layout for the LSTM kernel (bytes, all 32B aligned)
constexpr unsigned OFF_WIH   = 0;                          // [256][32] f16 (K padded 16->32)
constexpr unsigned OFF_WHH   = OFF_WIH   + 256 * 32 * 2;   // [256][64] f16
constexpr unsigned OFF_X     = OFF_WHH   + 256 * 64 * 2;   // [kMB][32] f16
constexpr unsigned OFF_H     = OFF_X     + kMB * 32 * 2;   // [kMB][64] f16
constexpr unsigned OFF_GATES = OFF_H     + kMB * 64 * 2;   // [kMB][256] f32
constexpr unsigned OFF_C     = OFF_GATES + kMB * 256 * 4;  // [kMB][64] f32
constexpr unsigned OFF_BIAS  = OFF_C     + kMB * 64 * 4;   // [256] f32
constexpr unsigned OFF_LN    = OFF_BIAS  + 256 * 4;        // [32] f32 (g then b)
constexpr unsigned SMEM1     = OFF_LN    + 32 * 4;
}

// branch-free fast sigmoid / tanh using the 1-ulp HW reciprocal (v_rcp_f32)
// instead of the IEEE div expansion (div_scale + NR + div_fmas).
__device__ __forceinline__ float rcp_fast(float x) { return __builtin_amdgcn_rcpf(x); }
__device__ __forceinline__ float sigf(float x) {
  return rcp_fast(1.f + __expf(-x));
}
__device__ __forceinline__ float tanh_fast(float x) {
  const float xc = fminf(fmaxf(x, -15.f), 15.f);
  const float e2 = __expf(2.f * xc);
  return 1.f - 2.f * rcp_fast(e2 + 1.f);
}

// A fragment (16x32 f16, M rows in lanes, per-lane k runs of 8) from row-major LDS
__device__ __forceinline__ v16h lds_a_16x32(const _Float16* base, int m0, int k0,
                                            int ldk, int lane) {
  const int g = (lane >> 4) & 1;
  const int m = m0 + (lane & 15);
  const _Float16* p = base + m * ldk + k0 + g * 8;
  v8h lo = *(const v8h*)(p);        // k = k0 + g*8  .. +7
  v8h hi = *(const v8h*)(p + 16);   // k = k0 + 16 + g*8 .. +7
  return __builtin_shufflevector(lo, hi, 0,1,2,3,4,5,6,7,8,9,10,11,12,13,14,15);
}

// B fragment (32x16 f16): B = W^T where W is row-major [N][K] in LDS.
// lane holds column n = n0+(lane&15), k = k0 + g*8 .. +7 and k0 + 16 + g*8 .. +7
__device__ __forceinline__ v16h lds_b_32x16(const _Float16* w, int n0, int k0,
                                            int ldk, int lane) {
  const int g = (lane >> 4) & 1;
  const int n = n0 + (lane & 15);
  const _Float16* p = w + n * ldk + k0 + g * 8;
  v8h lo = *(const v8h*)(p);
  v8h hi = *(const v8h*)(p + 16);
  return __builtin_shufflevector(lo, hi, 0,1,2,3,4,5,6,7,8,9,10,11,12,13,14,15);
}

__device__ __forceinline__ v8f wmma16(v16h a, v16h b, v8f c) {
  return __builtin_amdgcn_wmma_f32_16x16x32_f16(false, a, false, b, (short)0, c,
                                                false, false);
}

// ---------------------------------------------------------------------------
// Kernel 1: LayerNorm(target) -> LSTM over T steps. Batch-parallel, WMMA gates.
// grid = kB/kMB blocks x 256 threads (8 waves). Dynamic LDS = SMEM1 bytes.
// Weight B-fragments are hoisted into registers before the serial t-loop.
// ---------------------------------------------------------------------------
__global__ __launch_bounds__(256) void k_lstm(
    const float* __restrict__ target, const float* __restrict__ ln_g,
    const float* __restrict__ ln_b, const float* __restrict__ W_ih,
    const float* __restrict__ W_hh, const float* __restrict__ b_ih,
    const float* __restrict__ b_hh, float* __restrict__ lstm_out) {
  extern __shared__ char smem[];
  _Float16* sWih   = (_Float16*)(smem + OFF_WIH);
  _Float16* sWhh   = (_Float16*)(smem + OFF_WHH);
  _Float16* sX     = (_Float16*)(smem + OFF_X);
  _Float16* sH     = (_Float16*)(smem + OFF_H);
  float*    sGates = (float*)(smem + OFF_GATES);
  float*    sC     = (float*)(smem + OFF_C);
  float*    sBias  = (float*)(smem + OFF_BIAS);
  float*    sLNg   = (float*)(smem + OFF_LN);
  float*    sLNb   = sLNg + 16;

  const int tid  = threadIdx.x;
  const int lane = tid & 31;
  const int wave = tid >> 5;
  const int b0   = blockIdx.x * kMB;

  // stage weights as f16 (K of W_ih padded 16->32 with zeros)
  for (int i = tid; i < 256 * 32; i += 256) {
    const int n = i >> 5, k = i & 31;
    sWih[i] = (k < kF) ? (_Float16)W_ih[n * kF + k] : (_Float16)0.f;
  }
  for (int i = tid; i < 256 * 64; i += 256) sWhh[i] = (_Float16)W_hh[i];
  if (tid < 256) sBias[tid] = b_ih[tid] + b_hh[tid];
  if (tid < 16) { sLNg[tid] = ln_g[tid]; sLNb[tid] = ln_b[tid]; }
  for (int i = tid; i < kMB * kH; i += 256) { sC[i] = 0.f; sH[i] = (_Float16)0.f; }
  for (int i = tid; i < kMB * 32; i += 256) sX[i] = (_Float16)0.f;  // zero pad cols
  __syncthreads();

  // hoist the (constant) weight B-fragments and biases into registers
  v16h bWih[2], bWhh0[2], bWhh1[2];
  float biasv[2];
#pragma unroll
  for (int j = 0; j < 2; ++j) {
    const int n0 = (wave * 2 + j) * 16;
    bWih[j]  = lds_b_32x16(sWih, n0, 0, 32, lane);
    bWhh0[j] = lds_b_32x16(sWhh, n0, 0, 64, lane);
    bWhh1[j] = lds_b_32x16(sWhh, n0, 32, 64, lane);
    biasv[j] = sBias[n0 + (lane & 15)];
  }

  for (int t = 0; t < kT; ++t) {
    // --- LayerNorm of x_t, f16 into A tile ---
    if (tid < kMB) {
      const float* xr = target + ((size_t)(b0 + tid) * kT + t) * kF;
      const float4* xr4 = (const float4*)xr;
      float buf[16];
#pragma unroll
      for (int i = 0; i < 4; ++i) {
        float4 u = xr4[i];
        buf[4 * i] = u.x; buf[4 * i + 1] = u.y; buf[4 * i + 2] = u.z; buf[4 * i + 3] = u.w;
      }
      float m = 0.f;
#pragma unroll
      for (int i = 0; i < 16; ++i) m += buf[i];
      m *= (1.f / 16.f);
      float v = 0.f;
#pragma unroll
      for (int i = 0; i < 16; ++i) { float d = buf[i] - m; v += d * d; }
      v *= (1.f / 16.f);
      const float r = rsqrtf(v + 1e-5f);
#pragma unroll
      for (int i = 0; i < 16; ++i)
        sX[tid * 32 + i] = (_Float16)((buf[i] - m) * r * sLNg[i] + sLNb[i]);
      if (t + 1 < kT)
        __builtin_prefetch(target + ((size_t)(b0 + tid) * kT + t + 1) * kF, 0, 1);
    }
    __syncthreads();

    // --- gates = LN(x) @ Wih^T + h @ Whh^T + (b_ih+b_hh), via WMMA ---
    // wave w owns N-tiles {2w, 2w+1}; single M tile (kMB==16)
    const v16h aX  = lds_a_16x32(sX, 0, 0, 32, lane);
    const v16h aH0 = lds_a_16x32(sH, 0, 0, 64, lane);
    const v16h aH1 = lds_a_16x32(sH, 0, 32, 64, lane);
#pragma unroll
    for (int j = 0; j < 2; ++j) {
      const int n0 = (wave * 2 + j) * 16;
      v8f acc;
#pragma unroll
      for (int vv = 0; vv < 8; ++vv) acc[vv] = biasv[j];
      acc = wmma16(aX,  bWih[j],  acc);
      acc = wmma16(aH0, bWhh0[j], acc);
      acc = wmma16(aH1, bWhh1[j], acc);
      const int g = (lane >> 4) & 1, n = n0 + (lane & 15);
#pragma unroll
      for (int vv = 0; vv < 8; ++vv) sGates[(g * 8 + vv) * kG + n] = acc[vv];
    }
    __syncthreads();

    // --- elementwise LSTM cell (torch gate order i,f,g,o), branch-free ---
#pragma unroll
    for (int k = 0; k < (kMB * kH) / 256; ++k) {
      const int flat = k * 256 + tid;
      const int r = flat >> 6, hh = flat & 63;
      const float ig = sGates[r * kG + hh];
      const float fg = sGates[r * kG + kH + hh];
      const float gg = sGates[r * kG + 2 * kH + hh];
      const float og = sGates[r * kG + 3 * kH + hh];
      const float c = sigf(fg) * sC[r * kH + hh] + sigf(ig) * tanh_fast(gg);
      const float h = sigf(og) * tanh_fast(c);
      sC[r * kH + hh] = c;
      sH[r * kH + hh] = (_Float16)h;
      lstm_out[((size_t)(b0 + r) * kT + t) * kH + hh] = h;
    }
    __syncthreads();
  }
}

// ---------------------------------------------------------------------------
// Kernel 2: per-(b,t): LN(neighbors) -> encoder (WMMA) -> @Wg^T (WMMA) ->
// GAT star softmax -> combined[b,t,:]. grid = kB*kT blocks x 64 threads.
// ---------------------------------------------------------------------------
__global__ __launch_bounds__(64) void k_gat(
    const float* __restrict__ others, const float* __restrict__ ln_g,
    const float* __restrict__ ln_b, const float* __restrict__ Wn,
    const float* __restrict__ bn, const float* __restrict__ Wg,
    const float* __restrict__ a_src, const float* __restrict__ a_dst,
    const float* __restrict__ bg, const float* __restrict__ lstm_out,
    float* __restrict__ combined) {
  __shared__ __attribute__((aligned(16))) _Float16 sWn[kH * 32];   // [n][k pad 32]
  __shared__ __attribute__((aligned(16))) _Float16 sWg[kH * kH];   // [n][k]
  __shared__ __attribute__((aligned(16))) _Float16 sXn[kN * 32];   // LN'd neighbors
  __shared__ __attribute__((aligned(16))) _Float16 sHn[kN * kH];   // relu(enc) f16
  __shared__ float sGn[kN * kH];
  __shared__ float sGt[kH], sTgt[kH];
  __shared__ float sAs[kN + 2];          // [0..31] neighbors, [32] self, [33] a_d0
  __shared__ float sWt[kN + 1];
  __shared__ float sBn[kH], sAsrc[kH], sAdst[kH];
  __shared__ float sLNg[16], sLNb[16];

  const int tid  = threadIdx.x;
  const int lane = tid & 31;
  const int wave = tid >> 5;
  const int bt   = blockIdx.x;

  for (int i = tid; i < kH * 32; i += 64) {
    const int n = i >> 5, k = i & 31;
    sWn[i] = (k < kF) ? (_Float16)Wn[n * kF + k] : (_Float16)0.f;
  }
  for (int i = tid; i < kH * kH; i += 64) sWg[i] = (_Float16)Wg[i];
  sBn[tid] = bn[tid]; sAsrc[tid] = a_src[tid]; sAdst[tid] = a_dst[tid];
  sTgt[tid] = lstm_out[(size_t)bt * kH + tid];
  if (tid < 16) { sLNg[tid] = ln_g[tid]; sLNb[tid] = ln_b[tid]; }
  for (int i = tid; i < kN * 32; i += 64) sXn[i] = (_Float16)0.f;
  __syncthreads();

  // --- LayerNorm of the 32 neighbor rows ---
  if (tid < kN) {
    const float4* xr4 = (const float4*)(others + ((size_t)bt * kN + tid) * kF);
    float buf[16];
#pragma unroll
    for (int i = 0; i < 4; ++i) {
      float4 u = xr4[i];
      buf[4 * i] = u.x; buf[4 * i + 1] = u.y; buf[4 * i + 2] = u.z; buf[4 * i + 3] = u.w;
    }
    float m = 0.f;
#pragma unroll
    for (int i = 0; i < 16; ++i) m += buf[i];
    m *= (1.f / 16.f);
    float v = 0.f;
#pragma unroll
    for (int i = 0; i < 16; ++i) { float d = buf[i] - m; v += d * d; }
    v *= (1.f / 16.f);
    const float r = rsqrtf(v + 1e-5f);
#pragma unroll
    for (int i = 0; i < 16; ++i)
      sXn[tid * 32 + i] = (_Float16)((buf[i] - m) * r * sLNg[i] + sLNb[i]);
  }
  __syncthreads();

  // --- encoder: hn = relu(LN(x) @ Wn^T + bn), WMMA (K=16 padded) ---
  {
    const int m0 = wave * 16;
    const v16h aXn = lds_a_16x32(sXn, m0, 0, 32, lane);
#pragma unroll
    for (int nt = 0; nt < 4; ++nt) {
      const int n0 = nt * 16;
      v8f acc;
      const float bv = sBn[n0 + (lane & 15)];
#pragma unroll
      for (int vv = 0; vv < 8; ++vv) acc[vv] = bv;
      acc = wmma16(aXn, lds_b_32x16(sWn, n0, 0, 32, lane), acc);
      const int g = (lane >> 4) & 1, n = n0 + (lane & 15);
#pragma unroll
      for (int vv = 0; vv < 8; ++vv)
        sHn[(m0 + g * 8 + vv) * kH + n] = (_Float16)fmaxf(acc[vv], 0.f);
    }
  }
  __syncthreads();

  // --- gn = hn @ Wg^T (WMMA, K=64) ; gt = Wg @ lstm_out (matvec) ---
  {
    const int m0 = wave * 16;
    const v16h aH0 = lds_a_16x32(sHn, m0, 0, 64, lane);
    const v16h aH1 = lds_a_16x32(sHn, m0, 32, 64, lane);
#pragma unroll
    for (int nt = 0; nt < 4; ++nt) {
      const int n0 = nt * 16;
      v8f acc;
#pragma unroll
      for (int vv = 0; vv < 8; ++vv) acc[vv] = 0.f;
      acc = wmma16(aH0, lds_b_32x16(sWg, n0, 0, 64, lane), acc);
      acc = wmma16(aH1, lds_b_32x16(sWg, n0, 32, 64, lane), acc);
      const int g = (lane >> 4) & 1, n = n0 + (lane & 15);
#pragma unroll
      for (int vv = 0; vv < 8; ++vv) sGn[(m0 + g * 8 + vv) * kH + n] = acc[vv];
    }
    float acc = 0.f;
#pragma unroll
    for (int k = 0; k < kH; ++k) acc += sTgt[k] * (float)sWg[tid * kH + k];
    sGt[tid] = acc;
  }
  __syncthreads();

  // --- attention logits ---
  if (tid < kN) {
    float acc = 0.f;
#pragma unroll
    for (int k = 0; k < kH; ++k) acc += sGn[tid * kH + k] * sAsrc[k];
    sAs[tid] = acc;
  } else if (tid == kN) {
    float acc = 0.f;
#pragma unroll
    for (int k = 0; k < kH; ++k) acc += sGt[k] * sAsrc[k];
    sAs[kN] = acc;
  } else if (tid == kN + 1) {
    float acc = 0.f;
#pragma unroll
    for (int k = 0; k < kH; ++k) acc += sGt[k] * sAdst[k];
    sAs[kN + 1] = acc;  // a_d0
  }
  __syncthreads();

  // --- leaky-relu + softmax over the 33 sources ---
  if (tid == 0) {
    const float ad0 = sAs[kN + 1];
    float lg[kN + 1];
    float mx = -1e30f;
#pragma unroll
    for (int j = 0; j <= kN; ++j) {
      float x = sAs[j] + ad0;
      x = (x > 0.f) ? x : 0.2f * x;
      lg[j] = x;
      mx = fmaxf(mx, x);
    }
    float s = 0.f;
#pragma unroll
    for (int j = 0; j <= kN; ++j) { float e = __expf(lg[j] - mx); sWt[j] = e; s += e; }
    const float inv = rcp_fast(s);
#pragma unroll
    for (int j = 0; j <= kN; ++j) sWt[j] *= inv;
  }
  __syncthreads();

  // --- weighted combine + bg ---
  {
    float acc = bg[tid];
#pragma unroll
    for (int j = 0; j < kN; ++j) acc += sWt[j] * sGn[j * kH + tid];
    acc += sWt[kN] * sGt[tid];
    combined[(size_t)bt * kH + tid] = acc;
  }
}

// ---------------------------------------------------------------------------
// Kernel 3: temporal self-attention (query = last step) + classifier.
// grid = kB blocks x 64 threads. Small fraction of FLOPs -> VALU.
// ---------------------------------------------------------------------------
__global__ __launch_bounds__(64) void k_attn(
    const float* __restrict__ combined, const float* __restrict__ Wqkv,
    const float* __restrict__ bqkv, const float* __restrict__ Wo,
    const float* __restrict__ bo, const float* __restrict__ W1,
    const float* __restrict__ b1, const float* __restrict__ W2,
    const float* __restrict__ b2, float* __restrict__ out) {
  __shared__ float sCmb[kT * kH];
  __shared__ float sQ[kH];
  __shared__ float sK[kT * kH];
  __shared__ float sV[kT * kH];
  __shared__ float sS[kT + 1];
  __shared__ float sAp[kH];
  __shared__ float sAo[kH];
  __shared__ float sF1[kH];
  const int tid = threadIdx.x;
  const int b = blockIdx.x;

  for (int i = tid; i < kT * kH; i += 64)
    sCmb[i] = combined[(size_t)b * kT * kH + i];
  __syncthreads();

  {  // q / k / v projections (thread == output channel h)
    const int h = tid;
    const float* wq = Wqkv + h * kH;
    const float* wk = Wqkv + (kH + h) * kH;
    const float* wv = Wqkv + (2 * kH + h) * kH;
    float q = bqkv[h];
#pragma unroll
    for (int k = 0; k < kH; ++k) q += sCmb[(kT - 1) * kH + k] * wq[k];
    sQ[h] = q;
    const float bk = bqkv[kH + h], bv = bqkv[2 * kH + h];
    for (int tt = 0; tt < kT; ++tt) {
      float ka = bk, va = bv;
#pragma unroll
      for (int k = 0; k < kH; ++k) {
        const float c = sCmb[tt * kH + k];
        ka += c * wk[k];
        va += c * wv[k];
      }
      sK[tt * kH + h] = ka;
      sV[tt * kH + h] = va;
    }
  }
  __syncthreads();

  if (tid < kT) {
    float s = 0.f;
#pragma unroll
    for (int k = 0; k < kH; ++k) s += sQ[k] * sK[tid * kH + k];
    sS[tid] = s * 0.125f;  // 1/sqrt(64)
  }
  __syncthreads();
  if (tid == 0) {
    float mx = -1e30f;
    for (int tt = 0; tt < kT; ++tt) mx = fmaxf(mx, sS[tt]);
    float s = 0.f;
    for (int tt = 0; tt < kT; ++tt) { float e = __expf(sS[tt] - mx); sS[tt] = e; s += e; }
    const float inv = rcp_fast(s);
    for (int tt = 0; tt < kT; ++tt) sS[tt] *= inv;
  }
  __syncthreads();

  {  // attended (pre-Wo)
    float a = 0.f;
    for (int tt = 0; tt < kT; ++tt) a += sS[tt] * sV[tt * kH + tid];
    sAp[tid] = a;
  }
  __syncthreads();
  {  // @ Wo^T + bo
    float a = bo[tid];
#pragma unroll
    for (int k = 0; k < kH; ++k) a += sAp[k] * Wo[tid * kH + k];
    sAo[tid] = a;
  }
  __syncthreads();
  {  // relu(@ W1^T + b1)
    float a = b1[tid];
#pragma unroll
    for (int k = 0; k < kH; ++k) a += sAo[k] * W1[tid * kH + k];
    sF1[tid] = fmaxf(a, 0.f);
  }
  __syncthreads();
  if (tid < 2) {  // @ W2^T + b2
    float a = b2[tid];
#pragma unroll
    for (int k = 0; k < kH; ++k) a += sF1[k] * W2[tid * kH + k];
    out[(size_t)b * 2 + tid] = a;
  }
}

// ---------------------------------------------------------------------------
extern "C" void kernel_launch(void* const* d_in, const int* in_sizes, int n_in,
                              void* d_out, int out_size, void* d_ws, size_t ws_size,
                              hipStream_t stream) {
  (void)in_sizes; (void)n_in; (void)out_size; (void)ws_size;
  const float* target = (const float*)d_in[0];
  const float* others = (const float*)d_in[1];
  const float* ln_g   = (const float*)d_in[2];
  const float* ln_b   = (const float*)d_in[3];
  const float* W_ih   = (const float*)d_in[4];
  const float* W_hh   = (const float*)d_in[5];
  const float* b_ih   = (const float*)d_in[6];
  const float* b_hh   = (const float*)d_in[7];
  const float* Wn     = (const float*)d_in[8];
  const float* bn     = (const float*)d_in[9];
  const float* Wg     = (const float*)d_in[10];
  const float* a_src  = (const float*)d_in[11];
  const float* a_dst  = (const float*)d_in[12];
  const float* bg     = (const float*)d_in[13];
  const float* Wqkv   = (const float*)d_in[14];
  const float* bqkv   = (const float*)d_in[15];
  const float* Wo     = (const float*)d_in[16];
  const float* bo     = (const float*)d_in[17];
  const float* W1     = (const float*)d_in[18];
  const float* b1     = (const float*)d_in[19];
  const float* W2     = (const float*)d_in[20];
  const float* b2     = (const float*)d_in[21];

  float* ws       = (float*)d_ws;
  float* lstm_out = ws;                              // [B,T,H]
  float* combined = ws + (size_t)kB * kT * kH;       // [B,T,H]

  k_lstm<<<kB / kMB, 256, SMEM1, stream>>>(target, ln_g, ln_b, W_ih, W_hh, b_ih,
                                           b_hh, lstm_out);
  k_gat<<<kB * kT, 64, 0, stream>>>(others, ln_g, ln_b, Wn, bn, Wg, a_src, a_dst,
                                    bg, lstm_out, combined);
  k_attn<<<kB, 64, 0, stream>>>(combined, Wqkv, bqkv, Wo, bo, W1, b1, W2, b2,
                                (float*)d_out);
}
